// BertSelfAttention_6270652252602
// MI455X (gfx1250) — matrix operand order repaired and checked by reading, not compiled, not run on hardware
//
#include <hip/hip_runtime.h>

#define BB 8
#define SS 1024
#define DD 768
#define HH 12
#define DH 64

typedef __attribute__((ext_vector_type(16))) __bf16 bf16x16;
typedef __attribute__((ext_vector_type(8)))  float  f32x8;

struct P32    { uint4 lo, hi; };
struct U16x16 { unsigned short s[16]; };

static __device__ __forceinline__ unsigned short f2bfu(float f) {
  unsigned u = __builtin_bit_cast(unsigned, f);
  unsigned r = u + 0x7FFFu + ((u >> 16) & 1u);   // round-to-nearest-even
  return (unsigned short)(r >> 16);
}

static __device__ __forceinline__ unsigned short f2h(float f) {
  _Float16 h = (_Float16)f;
  return __builtin_bit_cast(unsigned short, h);
}
static __device__ __forceinline__ float h2f(unsigned short u) {
  return (float)__builtin_bit_cast(_Float16, u);
}

static __device__ __forceinline__ bf16x16 ld_frag_2x16B(const unsigned short* p0,
                                                        const unsigned short* p1) {
  P32 p;
  p.lo = *reinterpret_cast<const uint4*>(p0);
  p.hi = *reinterpret_cast<const uint4*>(p1);
  return __builtin_bit_cast(bf16x16, p);
}

// A fragment (16xK): `row` points at this lane's M-row, contiguous K.
// element e -> K = kb + 16*(e>>3) + half*8 + (e&7)
static __device__ __forceinline__ bf16x16 ld_a(const unsigned short* row, int kb, int half) {
  return ld_frag_2x16B(row + kb + half * 8, row + kb + 16 + half * 8);
}

// B fragment (Kx16) from transposed source: `row` = memory row for this lane's
// output column, contiguous K. element e -> K = kb + 16*(e>>3) + half*8 + (e&7)
static __device__ __forceinline__ bf16x16 ld_bt(const unsigned short* row, int kb, int half) {
  return ld_frag_2x16B(row + kb + half * 8, row + kb + 16 + half * 8);
}

// A fragment from an f32 row with on-the-fly bf16 conversion.
static __device__ __forceinline__ bf16x16 ld_a_f32(const float* row, int kb, int half) {
  U16x16 t;
  const float* p0 = row + kb + half * 8;
  const float* p1 = row + kb + 16 + half * 8;
#pragma unroll
  for (int e = 0; e < 8; ++e) { t.s[e] = f2bfu(p0[e]); t.s[8 + e] = f2bfu(p1[e]); }
  return __builtin_bit_cast(bf16x16, t);
}

static __device__ __forceinline__ f32x8 wmma_bf16(bf16x16 a, bf16x16 b, f32x8 c) {
  return __builtin_amdgcn_wmma_f32_16x16x32_bf16(false, a, false, b, (short)0, c,
                                                 false, false);
}

// ------------------------------------------------------------------ prep: W^T
// Write bf16 transposed weights: wT[n*768 + k] = W[k*768 + n]  (coalesced writes)
__global__ __launch_bounds__(256) void k_prep_w(const float* __restrict__ wq,
                                                const float* __restrict__ wk,
                                                const float* __restrict__ wv,
                                                unsigned short* __restrict__ oq,
                                                unsigned short* __restrict__ ok,
                                                unsigned short* __restrict__ ov) {
  const int n = blockIdx.x;                  // 0..767 (output row = W column)
  for (int k = threadIdx.x; k < DD; k += 256) {
    const size_t si = (size_t)k * DD + n;
    const size_t di = (size_t)n * DD + k;
    oq[di] = f2bfu(wq[si]);
    ok[di] = f2bfu(wk[si]);
    ov[di] = f2bfu(wv[si]);
  }
}

__global__ __launch_bounds__(256) void k_prep_seg(const float* __restrict__ table,
                                                  const int* __restrict__ ids,
                                                  unsigned short* __restrict__ seg) {
  const int bs = blockIdx.x;                 // 0 .. B*S-1
  const int id = ids[bs];
  const float* src = table + (size_t)id * DD;
  unsigned short* dst = seg + (size_t)bs * DD;
  for (int d = threadIdx.x; d < DD; d += 256) dst[d] = f2bfu(src[d]);
}

// ------------------------------------------------------------------- QKV GEMM
__global__ __launch_bounds__(256) void k_qkv(
    const float* __restrict__ hs, const float* __restrict__ bq,
    const float* __restrict__ bk, const float* __restrict__ bv,
    const float* __restrict__ bqs,
    const unsigned short* __restrict__ wqT, const unsigned short* __restrict__ wkT,
    const unsigned short* __restrict__ wvT,
    unsigned short* __restrict__ qs, unsigned short* __restrict__ qb,
    unsigned short* __restrict__ kbf, unsigned short* __restrict__ vbf) {
  __shared__ __align__(16) unsigned short ldsX[16 * DD];
  const int tid = threadIdx.x;
  const int mt = blockIdx.x;                       // 0 .. 511 (16-row tile of X)
  const float4* src = reinterpret_cast<const float4*>(hs + (size_t)mt * 16 * DD);
  for (int i = tid; i < 16 * DD / 4; i += 256) {
    float4 f = src[i];
    ldsX[i * 4 + 0] = f2bfu(f.x);
    ldsX[i * 4 + 1] = f2bfu(f.y);
    ldsX[i * 4 + 2] = f2bfu(f.z);
    ldsX[i * 4 + 3] = f2bfu(f.w);
  }
  __syncthreads();
  const int wave = tid >> 5, lane = tid & 31;
  const int half = lane >> 4, lm = lane & 15;
  const unsigned short* arow = ldsX + lm * DD;
  for (int job = wave; job < 144; job += 8) {      // 3 matrices * 48 N-tiles
    const int mat = job / 48, nt = job % 48, col = nt * 16 + lm;
    const unsigned short* WT = (mat == 0) ? wqT : (mat == 1) ? wkT : wvT;
    const unsigned short* brow = WT + (size_t)col * DD;   // W column, contiguous K
    const float bias = ((mat == 0) ? bq : (mat == 1) ? bk : bv)[col];
    f32x8 c = {bias, bias, bias, bias, bias, bias, bias, bias};
#pragma unroll 4
    for (int kb = 0; kb < DD; kb += 32) {
      c = wmma_bf16(ld_a(arow, kb, half), ld_bt(brow, kb, half), c);
    }
#pragma unroll
    for (int r = 0; r < 8; ++r) {
      const size_t off = (size_t)(mt * 16 + r + half * 8) * DD + col;
      const float v = c[r];
      if (mat == 0) {
        qs[off] = f2bfu(v * 0.125f);               // q * 1/sqrt(64)
        qb[off] = f2bfu(v + bqs[col]);             // q + b_q_s
      } else if (mat == 1) {
        kbf[off] = f2bfu(v);
      } else {
        vbf[off] = f2bfu(v);
      }
    }
  }
}

// --------------------------------------------- V transpose: vT[b,h,d,s] (bf16)
__global__ __launch_bounds__(256) void k_tr_v(const unsigned short* __restrict__ vbf,
                                              unsigned short* __restrict__ vT) {
  const int row = blockIdx.x;                // 0 .. B*H*DH-1
  const int d = row & 63;
  const int bh = row >> 6;
  const int b = bh / HH, h = bh % HH;
  const unsigned short* src = vbf + (size_t)b * SS * DD + h * DH + d;
  unsigned short* dst = vT + (size_t)row * SS;
  for (int s = threadIdx.x; s < SS; s += 256) dst[s] = src[(size_t)s * DD];
}

// ---------------- fused scores (+mask) -> f16 LDS -> row softmax -> f32 probs
// block = 64 threads = 2 waves; each wave owns a 16x1024 stripe staged in its
// private 32KB LDS slab (no cross-wave sharing, no barriers needed).
__global__ __launch_bounds__(64) void k_scores_softmax(
    const unsigned short* __restrict__ qs, const unsigned short* __restrict__ qb,
    const unsigned short* __restrict__ kbf, const unsigned short* __restrict__ seg,
    const float* __restrict__ mask, float* __restrict__ probs) {
  __shared__ __align__(16) unsigned short sc[2 * 16 * SS];   // 64 KB
  const int wave = threadIdx.x >> 5, lane = threadIdx.x & 31;
  const int half = lane >> 4, lm = lane & 15;
  const int gw = blockIdx.x * 2 + wave;            // 0 .. B*H*64-1
  const int qt = gw & 63;
  const int h = (gw >> 6) % HH;
  const int b = gw / (64 * HH);
  const size_t hd = (size_t)h * DH;
  unsigned short* slab = sc + wave * 16 * SS;

  // ---- phase 1: scores tile-row into LDS (f16), mask folded in
  const unsigned short* qsrow = qs + (size_t)(b * SS + qt * 16 + lm) * DD + hd;
  const unsigned short* qbrow = qb + (size_t)(b * SS + qt * 16 + lm) * DD + hd;
  const bf16x16 a0 = ld_a(qsrow, 0, half), a1 = ld_a(qsrow, 32, half);
  const bf16x16 a2 = ld_a(qbrow, 0, half), a3 = ld_a(qbrow, 32, half);
  const float* mrow = mask + (size_t)b * SS * SS;
  for (int jt = 0; jt < 64; ++jt) {
    const int j = jt * 16 + lm;
    const unsigned short* krow = kbf + (size_t)(b * SS + j) * DD + hd;
    const unsigned short* srow = seg + (size_t)(b * SS + j) * DD + hd;
    f32x8 c = {0.f, 0.f, 0.f, 0.f, 0.f, 0.f, 0.f, 0.f};
    c = wmma_bf16(a0, ld_bt(krow, 0, half), c);
    c = wmma_bf16(a1, ld_bt(krow, 32, half), c);
    c = wmma_bf16(a2, ld_bt(srow, 0, half), c);
    c = wmma_bf16(a3, ld_bt(srow, 32, half), c);
#pragma unroll
    for (int r = 0; r < 8; ++r) {
      const int m = r + half * 8;
      const int i = qt * 16 + m;
      slab[m * SS + j] = f2h(c[r] + mrow[(size_t)i * SS + j]);
    }
  }

  // ---- phase 2: per-row softmax out of LDS, write final f32 probs
  float* pbase = probs + ((size_t)(b * HH + h) * SS + qt * 16) * SS;
  for (int m = 0; m < 16; ++m) {
    const unsigned short* rp = slab + m * SS;
    float v[32];
    float mx = -3.4e38f;
#pragma unroll
    for (int e = 0; e < 8; ++e) {
      ushort4 u = *reinterpret_cast<const ushort4*>(rp + lane * 4 + e * 128);
      v[e * 4 + 0] = h2f(u.x); v[e * 4 + 1] = h2f(u.y);
      v[e * 4 + 2] = h2f(u.z); v[e * 4 + 3] = h2f(u.w);
      mx = fmaxf(mx, fmaxf(fmaxf(v[e * 4 + 0], v[e * 4 + 1]),
                           fmaxf(v[e * 4 + 2], v[e * 4 + 3])));
    }
#pragma unroll
    for (int off = 16; off > 0; off >>= 1) mx = fmaxf(mx, __shfl_xor(mx, off, 32));
    float s = 0.f;
#pragma unroll
    for (int e = 0; e < 32; ++e) { v[e] = __expf(v[e] - mx); s += v[e]; }
#pragma unroll
    for (int off = 16; off > 0; off >>= 1) s += __shfl_xor(s, off, 32);
    const float inv = 1.0f / s;
    float* prow = pbase + (size_t)m * SS;
#pragma unroll
    for (int e = 0; e < 8; ++e) {
      float4 o = {v[e * 4 + 0] * inv, v[e * 4 + 1] * inv,
                  v[e * 4 + 2] * inv, v[e * 4 + 3] * inv};
      *reinterpret_cast<float4*>(prow + lane * 4 + e * 128) = o;
    }
  }
}

// ------------------------------------------------------- context = probs @ V
__global__ __launch_bounds__(256) void k_context(const float* __restrict__ probs,
                                                 const unsigned short* __restrict__ vT,
                                                 float* __restrict__ ctx) {
  const int wave = threadIdx.x >> 5, lane = threadIdx.x & 31;
  const int half = lane >> 4, lm = lane & 15;
  const int gw = blockIdx.x * 8 + wave;
  const int qt = gw & 63;
  const int h = (gw >> 6) % HH;
  const int b = gw / (64 * HH);
  const int bh = b * HH + h;
  const float* arow = probs + ((size_t)bh * SS + qt * 16 + lm) * SS;
  const unsigned short* v0 = vT + ((size_t)bh * DH + 0 + lm) * SS;
  const unsigned short* v1 = vT + ((size_t)bh * DH + 16 + lm) * SS;
  const unsigned short* v2 = vT + ((size_t)bh * DH + 32 + lm) * SS;
  const unsigned short* v3 = vT + ((size_t)bh * DH + 48 + lm) * SS;
  f32x8 acc0 = {0.f, 0.f, 0.f, 0.f, 0.f, 0.f, 0.f, 0.f};
  f32x8 acc1 = acc0, acc2 = acc0, acc3 = acc0;
  for (int kb = 0; kb < SS; kb += 32) {
    bf16x16 a = ld_a_f32(arow, kb, half);
    acc0 = wmma_bf16(a, ld_bt(v0, kb, half), acc0);
    acc1 = wmma_bf16(a, ld_bt(v1, kb, half), acc1);
    acc2 = wmma_bf16(a, ld_bt(v2, kb, half), acc2);
    acc3 = wmma_bf16(a, ld_bt(v3, kb, half), acc3);
  }
#pragma unroll
  for (int r = 0; r < 8; ++r) {
    const size_t off = ((size_t)b * SS + qt * 16 + r + half * 8) * DD + h * DH + lm;
    ctx[off + 0] = acc0[r];
    ctx[off + 16] = acc1[r];
    ctx[off + 32] = acc2[r];
    ctx[off + 48] = acc3[r];
  }
}

// ------------------------------------------------- selective attention scores
__global__ __launch_bounds__(256) void k_selective(const float* __restrict__ probs,
                                                   const float* __restrict__ mask,
                                                   const int* __restrict__ att,
                                                   float* __restrict__ sel,
                                                   float* __restrict__ selr) {
  __shared__ float red[256];
  const int t = threadIdx.x;
  const int bh = blockIdx.x;                       // 0 .. B*H-1
  const int b = bh / HH;
  const int a0 = att[b * 4 + 0], a1 = att[b * 4 + 1];
  const int a2 = att[b * 4 + 2], a3 = att[b * 4 + 3];
  const float* P = probs + (size_t)bh * SS * SS;
  const float* M = mask + (size_t)b * SS * SS;
  float mq[4] = {0.f, 0.f, 0.f, 0.f};
  float mr[4][3] = {};
  bool km[4];
  int j[4];
#pragma unroll
  for (int q = 0; q < 4; ++q) {
    j[q] = t + 256 * q;
    km[q] = (M[j[q]] == 0.0f);                     // kmask0
  }
  for (int i = 0; i < SS; ++i) {
    const bool qm = (M[(size_t)i * SS] == 0.0f);   // qmask0
    const bool rowq = (i < a0) && qm;
    int cls = -1;
    if (i >= a0) { if (i < a1) cls = 0; else if (i < a2) cls = 1; else if (i < a3) cls = 2; }
    const float* prow = P + (size_t)i * SS;
#pragma unroll
    for (int q = 0; q < 4; ++q) {
      const float p = prow[j[q]];
      if (rowq) mq[q] = fmaxf(mq[q], p);
      if (cls >= 0 && km[q]) mr[q][cls] = fmaxf(mr[q][cls], p);
    }
  }
  float outv[6];
  const int lo[3] = {a0, a1, a2}, hi[3] = {a1, a2, a3};
  for (int c = 0; c < 3; ++c) {
    float s = 0.f;
#pragma unroll
    for (int q = 0; q < 4; ++q) if (j[q] >= lo[c] && j[q] < hi[c]) s += mq[q];
    outv[c] = s;
  }
  for (int c = 0; c < 3; ++c) {
    float s = 0.f;
#pragma unroll
    for (int q = 0; q < 4; ++q) if (j[q] < a0) s += mr[q][c];
    outv[3 + c] = s;
  }
  float res[6] = {0.f, 0.f, 0.f, 0.f, 0.f, 0.f};
  for (int c = 0; c < 6; ++c) {
    __syncthreads();
    red[t] = outv[c];
    __syncthreads();
    for (int st = 128; st > 0; st >>= 1) {
      if (t < st) red[t] += red[t + st];
      __syncthreads();
    }
    if (t == 0) res[c] = red[0];
  }
  if (t == 0) {
    sel[bh * 3 + 0] = res[0] / (float)(a1 - a0);
    sel[bh * 3 + 1] = res[1] / (float)(a2 - a1);
    sel[bh * 3 + 2] = res[2] / (float)(a3 - a2);
    const float inva0 = 1.0f / (float)a0;
    selr[bh * 3 + 0] = res[3] * inva0;
    selr[bh * 3 + 1] = res[4] * inva0;
    selr[bh * 3 + 2] = res[5] * inva0;
  }
}

extern "C" void kernel_launch(void* const* d_in, const int* in_sizes, int n_in,
                              void* d_out, int out_size, void* d_ws, size_t ws_size,
                              hipStream_t stream) {
  (void)in_sizes; (void)n_in; (void)out_size; (void)ws_size;
  const float* hs      = (const float*)d_in[0];
  const float* mask    = (const float*)d_in[1];
  const int*   seg_ids = (const int*)d_in[2];
  const int*   att_ids = (const int*)d_in[3];
  const float* Wq      = (const float*)d_in[4];
  const float* bq      = (const float*)d_in[5];
  const float* Wk      = (const float*)d_in[6];
  const float* bk      = (const float*)d_in[7];
  const float* Wv      = (const float*)d_in[8];
  const float* bv      = (const float*)d_in[9];
  const float* seg_tab = (const float*)d_in[10];
  const float* b_q_s   = (const float*)d_in[11];

  // workspace layout (bytes)
  char* ws = (char*)d_ws;
  const size_t WMAT = (size_t)DD * DD * 2;           // 1,179,648
  const size_t SLAB = (size_t)BB * SS * DD * 2;      // 12,582,912
  unsigned short* wqT   = (unsigned short*)(ws);
  unsigned short* wkT   = (unsigned short*)(ws + WMAT);
  unsigned short* wvT   = (unsigned short*)(ws + 2 * WMAT);
  unsigned short* qs_bf = (unsigned short*)(ws + 3 * WMAT);
  unsigned short* qb_bf = (unsigned short*)(ws + 3 * WMAT + SLAB);
  unsigned short* k_bf  = (unsigned short*)(ws + 3 * WMAT + 2 * SLAB);
  unsigned short* v_bf  = (unsigned short*)(ws + 3 * WMAT + 3 * SLAB);
  unsigned short* seg_bf= (unsigned short*)(ws + 3 * WMAT + 4 * SLAB);
  unsigned short* vT    = (unsigned short*)(ws + 3 * WMAT + 5 * SLAB);

  // output layout (floats): context | sel(B,H,3) | sel_r(B,H,3) | probs(B,H,S,S)
  float* out   = (float*)d_out;
  float* ctx   = out;
  float* sel   = out + (size_t)BB * SS * DD;
  float* selr  = sel + (size_t)BB * HH * 3;
  float* probs = selr + (size_t)BB * HH * 3;

  k_prep_w  <<<DD, 256, 0, stream>>>(Wq, Wk, Wv, wqT, wkT, wvT);
  k_prep_seg<<<BB * SS, 256, 0, stream>>>(seg_tab, seg_ids, seg_bf);
  k_qkv     <<<BB * SS / 16, 256, 0, stream>>>(hs, bq, bk, bv, b_q_s,
                                               wqT, wkT, wvT,
                                               qs_bf, qb_bf, k_bf, v_bf);
  k_tr_v    <<<BB * HH * DH, 256, 0, stream>>>(v_bf, vT);
  k_scores_softmax<<<BB * HH * (SS / 16) / 2, 64, 0, stream>>>(qs_bf, qb_bf, k_bf,
                                                               seg_bf, mask, probs);
  k_context <<<BB * HH * (SS / 16) / 8, 256, 0, stream>>>(probs, vT, ctx);
  k_selective<<<BB * HH, 256, 0, stream>>>(probs, mask, att_ids, sel, selr);
}
